// BasisExpertsAttention_36163624632751
// MI455X (gfx1250) — matrix operand order repaired and checked by reading, not compiled, not run on hardware
//
#include <hip/hip_runtime.h>
#include <hip/hip_bf16.h>
#include <math.h>

// ============================================================================
// BasisExpertsAttention for MI455X (gfx1250, wave32, WMMA).
//
// All heavy math runs through one templated WMMA GEMM kernel using
// v_wmma_f32_16x16x32_f16 (f16 inputs, f32 accumulate) with fused epilogues:
//   - bias + sin(30x)            (SIREN layers)
//   - bias + tanh-GELU           (FF up-projection)
//   - bias + residual f32 add    (attention O-proj, FF down-projection, K-split)
//   - bias + top-2 MoE mix into hv (expert final layer, folds view_proj)
// Softmax / layernorm / top-k / 4x4-per-head attention are VALU kernels
// (negligible FLOPs). Weights are converted once per launch to f16,
// transposed to [N][K] (K zero-padded to a multiple of 32) so both A and B
// WMMA fragments are contiguous 8-half (b128) global loads per the CDNA5
// fragment layouts.
//
// Input index map (JAX pytree flattening = sorted dict keys):
//   0 coords | 1 ctx_token
//   2..25  decoders[i]: W0(512,256) b0 W1(256,256) b1 W2 b2 W3(256,od) b3
//   26..33 experts: W0(8,39,256) b0(8,256) W1(8,256,256) b1 W2 b2 W3 b3
//   34..65 fusion[L] (ln1 ln2 w1 w2 wk wo wq wv): g/b or W/b pairs
//   66..73 gate: W0(19,128) b0 W1(128,128) b1 W2 b2 W3(128,8) b3
//   74 view_embedding(3,16) | 75 view_proj(16,256)
// ============================================================================

typedef _Float16 half_t;
typedef __attribute__((ext_vector_type(16))) _Float16 v16h;
typedef __attribute__((ext_vector_type(8)))  _Float16 v8h;
typedef __attribute__((ext_vector_type(8)))  float    v8f;

#define NPTS   32768
#define NVIEW  3
#define NEXP   8
#define DMODEL 256
#define W0SIREN 30.0f

enum { ACT_NONE = 0, ACT_SIN = 1, ACT_GELU = 2 };
enum { EPI_F16 = 0, EPI_RESID = 1, EPI_MIX = 2 };

// ---------------------------------------------------------------------------
// Core WMMA GEMM: C[M x Nout] = act(A[M x K](f16) * Bt[Nout x K](f16)^T + b)
// Block = 256 threads = 8 waves; wave (wr,wc) owns 16x16 tile of a 32x64
// block tile. K-loop step 32 with v_wmma_f32_16x16x32_f16.
// Fragment layouts (CDNA5 ISA 7.12.2):
//   A 16x32 f16 : lane<16 -> row=lane, K {k0..k0+7, k0+16..k0+23}
//                 lane>=16 -> row=lane-16, K {k0+8..15, k0+24..31}
//   B 32x16 f16 : lane<16 -> col=lane, K {k0..k0+7, k0+16..k0+23} ; lane>=16 -> K {k0+8..15, k0+24..31}
//   C f32       : vgpr r, lane l -> row = r + (l<16?0:8), col = l%16
// ---------------------------------------------------------------------------
template<int ACT, int EPI>
__global__ __launch_bounds__(256) void gemm_kernel(
    const half_t* __restrict__ A, int lda,
    const half_t* __restrict__ Bt, int ldb,
    const float* __restrict__ bias, int K,
    half_t* __restrict__ out16, int ldo,
    float* __restrict__ out32, const float* __restrict__ resid, int addbias,
    const float* __restrict__ wg, const float* __restrict__ vp,
    float* __restrict__ hv, int e, int Mrows)
{
    const int tid  = threadIdx.x;
    const int wave = tid >> 5;
    const int lane = tid & 31;
    const int hsel = lane >> 4;
    const int ln   = lane & 15;

    const int rowbase = blockIdx.x * 32 + (wave >> 2) * 16;
    const int colbase = blockIdx.y * 64 + (wave & 3) * 16;

    const half_t* ap = A  + (size_t)(rowbase + ln) * lda + hsel * 8;
    const half_t* bp = Bt + (size_t)(colbase + ln) * ldb + hsel * 8;

    v8f c = {};
    for (int k0 = 0; k0 < K; k0 += 32) {
        v8h alo = *(const v8h*)(ap + k0);
        v8h ahi = *(const v8h*)(ap + k0 + 16);
        v8h blo = *(const v8h*)(bp + k0);
        v8h bhi = *(const v8h*)(bp + k0 + 16);
        v16h a = __builtin_shufflevector(alo, ahi, 0,1,2,3,4,5,6,7,8,9,10,11,12,13,14,15);
        v16h b = __builtin_shufflevector(blo, bhi, 0,1,2,3,4,5,6,7,8,9,10,11,12,13,14,15);
        c = __builtin_amdgcn_wmma_f32_16x16x32_f16(false, a, false, b, (short)0, c, false, false);
    }

    const int col = colbase + ln;
#pragma unroll
    for (int r = 0; r < 8; ++r) {
        const int row = rowbase + r + hsel * 8;
        float val = c[r];
        if (addbias) val += bias[col];
        if (ACT == ACT_SIN) {
            val = sinf(W0SIREN * val);
        } else if (ACT == ACT_GELU) {
            float x3 = val * val * val;
            val = 0.5f * val * (1.0f + tanhf(0.7978845608028654f * (val + 0.044715f * x3)));
        }
        if (EPI == EPI_F16) {
            out16[(size_t)row * ldo + col] = (half_t)val;
        } else if (EPI == EPI_RESID) {
            size_t o = (size_t)row * DMODEL + col;
            out32[o] = resid[o] + val;
        } else { // EPI_MIX: hv[v][row][col] (+)= wg[v][row][e]*val (+ vp at e==0)
#pragma unroll
            for (int vv = 0; vv < NVIEW; ++vv) {
                float w  = wg[((size_t)vv * Mrows + row) * NEXP + e];
                size_t o = ((size_t)vv * Mrows + row) * DMODEL + col;
                if (e == 0) hv[o] = w * val + vp[vv * DMODEL + col];
                else        hv[o] += w * val;
            }
        }
    }
}

// --------------------------- support kernels -------------------------------

// Weight prep: src f32 [K][N] row-major -> dst f16 [N][Kp], K-padded w/ zeros.
__global__ void prep_kernel(const float* __restrict__ src, half_t* __restrict__ dst,
                            int K, int N, int Kp)
{
    int id = blockIdx.x * 256 + threadIdx.x;
    if (id >= N * Kp) return;
    int n = id / Kp, kp = id % Kp;
    dst[id] = (kp < K) ? (half_t)src[(size_t)kp * N + n] : (half_t)0.f;
}

// Positional encoding: [x, sin(x_i*2^k*pi), cos(...)] -> f16 [NPTS][64] (pad 39->64)
__global__ void pe_kernel(const float* __restrict__ coords, half_t* __restrict__ pe)
{
    int id = blockIdx.x * 256 + threadIdx.x; // NPTS*64
    int row = id >> 6, j = id & 63;
    float val = 0.f;
    if (j < 3) {
        val = coords[row * 3 + j];
    } else if (j < 21) {
        int t = j - 3, i = t / 6, k = t % 6;
        val = sinf(coords[row * 3 + i] * ((float)(1 << k) * 3.14159265358979323846f));
    } else if (j < 39) {
        int t = j - 21, i = t / 6, k = t % 6;
        val = cosf(coords[row * 3 + i] * ((float)(1 << k) * 3.14159265358979323846f));
    }
    pe[id] = (half_t)val;
}

// Gate input: [coords(3), view_emb(16)] -> f16 [V*NPTS][32] (pad 19->32)
__global__ void gin_kernel(const float* __restrict__ coords, const float* __restrict__ ve,
                           half_t* __restrict__ gin)
{
    int id = blockIdx.x * 256 + threadIdx.x; // VN*32
    int row = id >> 5, j = id & 31;
    int v = row / NPTS, n = row % NPTS;
    float val = 0.f;
    if (j < 3) val = coords[n * 3 + j];
    else if (j < 19) val = ve[v * 16 + (j - 3)];
    gin[id] = (half_t)val;
}

// Gate final layer + softmax + top-2 + renormalize -> wg f32 [V][NPTS][8]
__global__ __launch_bounds__(128) void gate_final_kernel(
    const half_t* __restrict__ h, const float* __restrict__ W,
    const float* __restrict__ bb, float* __restrict__ wg, int rows)
{
    __shared__ float sw[128 * 8 + 8];
    for (int t = threadIdx.x; t < 128 * 8; t += 128) sw[t] = W[t];
    if (threadIdx.x < 8) sw[1024 + threadIdx.x] = bb[threadIdx.x];
    __syncthreads();
    int r = blockIdx.x * 128 + threadIdx.x;
    if (r >= rows) return;
    const half_t* hp = h + (size_t)r * 128;
    float acc[8];
#pragma unroll
    for (int e2 = 0; e2 < 8; ++e2) acc[e2] = sw[1024 + e2];
    for (int kk = 0; kk < 128; ++kk) {
        float x = (float)hp[kk];
#pragma unroll
        for (int e2 = 0; e2 < 8; ++e2) acc[e2] += x * sw[kk * 8 + e2];
    }
    float m = acc[0];
#pragma unroll
    for (int e2 = 1; e2 < 8; ++e2) m = fmaxf(m, acc[e2]);
    float p[8], sum = 0.f;
#pragma unroll
    for (int e2 = 0; e2 < 8; ++e2) { p[e2] = expf(acc[e2] - m); sum += p[e2]; }
    float inv = 1.f / sum;
#pragma unroll
    for (int e2 = 0; e2 < 8; ++e2) p[e2] *= inv;
    int i1 = 0; float b1v = p[0];
#pragma unroll
    for (int e2 = 1; e2 < 8; ++e2) if (p[e2] > b1v) { b1v = p[e2]; i1 = e2; }
    int i2 = -1; float b2v = -1e30f;
#pragma unroll
    for (int e2 = 0; e2 < 8; ++e2) { if (e2 == i1) continue; if (p[e2] > b2v) { b2v = p[e2]; i2 = e2; } }
    float denom = 1.f / (b1v + b2v + 1e-9f);
#pragma unroll
    for (int e2 = 0; e2 < 8; ++e2)
        wg[(size_t)r * 8 + e2] = (e2 == i1 || e2 == i2) ? p[e2] * denom : 0.f;
}

// vp[v][c] = view_embedding[v] @ view_proj
__global__ void vp_kernel(const float* __restrict__ ve, const float* __restrict__ proj,
                          float* __restrict__ vp)
{
    int id = blockIdx.x * 256 + threadIdx.x;
    if (id >= NVIEW * DMODEL) return;
    int v = id / DMODEL, c = id % DMODEL;
    float acc = 0.f;
#pragma unroll
    for (int j = 0; j < 16; ++j) acc += ve[v * 16 + j] * proj[j * DMODEL + c];
    vp[id] = acc;
}

// tokens f32 [NPTS][4][256]: t0=ctx, t1+v = hv[v][n]
__global__ void tokens_kernel(const float* __restrict__ hv, const float* __restrict__ ctx,
                              float* __restrict__ tok)
{
    size_t id = (size_t)blockIdx.x * 256 + threadIdx.x;
    int c = (int)(id & 255);
    size_t row = id >> 8;
    int t = (int)(row & 3);
    size_t n = row >> 2;
    tok[id] = (t == 0) ? ctx[c]
                       : hv[(((size_t)(t - 1)) * NPTS + n) * DMODEL + c];
}

// LayerNorm: one wave per row of 256, f32 in -> f16 out
__global__ __launch_bounds__(256) void ln_kernel(
    const float* __restrict__ x, const float* __restrict__ g,
    const float* __restrict__ b, half_t* __restrict__ y, int rows)
{
    int wave = threadIdx.x >> 5, lane = threadIdx.x & 31;
    int row = blockIdx.x * 8 + wave;
    if (row >= rows) return;
    const float* xp = x + (size_t)row * DMODEL + lane * 8;
    float v0[8];
#pragma unroll
    for (int j = 0; j < 8; ++j) v0[j] = xp[j];
    float s = 0.f;
#pragma unroll
    for (int j = 0; j < 8; ++j) s += v0[j];
#pragma unroll
    for (int off = 16; off; off >>= 1) s += __shfl_xor(s, off);
    float mean = s * (1.0f / 256.0f);
    float q = 0.f;
#pragma unroll
    for (int j = 0; j < 8; ++j) { float d = v0[j] - mean; q += d * d; }
#pragma unroll
    for (int off = 16; off; off >>= 1) q += __shfl_xor(q, off);
    float inv = rsqrtf(q * (1.0f / 256.0f) + 1e-5f);
    half_t* yp = y + (size_t)row * DMODEL + lane * 8;
#pragma unroll
    for (int j = 0; j < 8; ++j) {
        int c = lane * 8 + j;
        yp[j] = (half_t)((v0[j] - mean) * inv * g[c] + b[c]);
    }
}

// Attention: one thread per (point n, head h, query token i); S=4, HD=32.
__global__ __launch_bounds__(256) void attn_kernel(
    const half_t* __restrict__ q, const half_t* __restrict__ k,
    const half_t* __restrict__ v, half_t* __restrict__ o)
{
    int id = blockIdx.x * 256 + threadIdx.x; // NPTS*8*4
    int n = id >> 5;
    int h = (id >> 2) & 7;
    int i = id & 3;
    if (n >= NPTS) return;
    const half_t* qp = q + ((size_t)(n * 4 + i)) * DMODEL + h * 32;
    float qv[32];
#pragma unroll
    for (int d = 0; d < 32; ++d) qv[d] = (float)qp[d];
    float s[4];
#pragma unroll
    for (int j = 0; j < 4; ++j) {
        const half_t* kp = k + ((size_t)(n * 4 + j)) * DMODEL + h * 32;
        float acc = 0.f;
#pragma unroll
        for (int d = 0; d < 32; ++d) acc += qv[d] * (float)kp[d];
        s[j] = acc * 0.17677669529663687f; // 1/sqrt(32)
    }
    float m = fmaxf(fmaxf(s[0], s[1]), fmaxf(s[2], s[3]));
    float sum = 0.f;
#pragma unroll
    for (int j = 0; j < 4; ++j) { s[j] = expf(s[j] - m); sum += s[j]; }
    float inv = 1.f / sum;
    float ov[32];
#pragma unroll
    for (int d = 0; d < 32; ++d) ov[d] = 0.f;
#pragma unroll
    for (int j = 0; j < 4; ++j) {
        const half_t* vpt = v + ((size_t)(n * 4 + j)) * DMODEL + h * 32;
        float a = s[j] * inv;
#pragma unroll
        for (int d = 0; d < 32; ++d) ov[d] += a * (float)vpt[d];
    }
    half_t* op = o + ((size_t)(n * 4 + i)) * DMODEL + h * 32;
#pragma unroll
    for (int d = 0; d < 32; ++d) op[d] = (half_t)ov[d];
}

// Decoder input: concat(tokens[n][1+i], tokens[n][0]) -> f16 [NPTS][512]
__global__ void decin_kernel(const float* __restrict__ tok, half_t* __restrict__ din, int i)
{
    size_t id = (size_t)blockIdx.x * 256 + threadIdx.x; // NPTS*512
    int j = (int)(id & 511);
    size_t n = id >> 9;
    float val = (j < 256) ? tok[((size_t)(n * 4 + 1 + i)) * DMODEL + j]
                          : tok[((size_t)(n * 4)) * DMODEL + (j - 256)];
    din[id] = (half_t)val;
}

// Decoder final: [NPTS][256] f16 x W(256,OD) f32 + b -> d_out[:, off..off+OD)
template<int OD>
__global__ __launch_bounds__(256) void dec_final_kernel(
    const half_t* __restrict__ h, const float* __restrict__ W,
    const float* __restrict__ bb, float* __restrict__ out, int outoff, int rows)
{
    __shared__ float sw[256 * OD + OD];
    for (int t = threadIdx.x; t < 256 * OD; t += 256) sw[t] = W[t];
    if (threadIdx.x < OD) sw[256 * OD + threadIdx.x] = bb[threadIdx.x];
    __syncthreads();
    int r = blockIdx.x * 256 + threadIdx.x;
    if (r >= rows) return;
    const half_t* hp = h + (size_t)r * 256;
    float acc[OD];
#pragma unroll
    for (int o2 = 0; o2 < OD; ++o2) acc[o2] = sw[256 * OD + o2];
    for (int kk = 0; kk < 256; ++kk) {
        float x = (float)hp[kk];
#pragma unroll
        for (int o2 = 0; o2 < OD; ++o2) acc[o2] += x * sw[kk * OD + o2];
    }
    float* op = out + (size_t)r * 20 + outoff;
#pragma unroll
    for (int o2 = 0; o2 < OD; ++o2) op[o2] = acc[o2];
}

// --------------------------- host-side helpers -----------------------------

template<int ACT>
static void gemm_f16(hipStream_t s, const half_t* A, int lda, const half_t* Bt, int ldb,
                     const float* bias, int K, int M, int N, half_t* out, int ldo)
{
    dim3 g(M / 32, N / 64), b(256);
    gemm_kernel<ACT, EPI_F16><<<g, b, 0, s>>>(A, lda, Bt, ldb, bias, K, out, ldo,
                                              nullptr, nullptr, 1,
                                              nullptr, nullptr, nullptr, 0, 0);
}

static void gemm_resid(hipStream_t s, const half_t* A, int lda, const half_t* Bt, int ldb,
                       const float* bias, int K, int M, float* out32, const float* resid,
                       int addbias)
{
    dim3 g(M / 32, DMODEL / 64), b(256);
    gemm_kernel<ACT_NONE, EPI_RESID><<<g, b, 0, s>>>(A, lda, Bt, ldb, bias, K,
                                                     nullptr, 0, out32, resid, addbias,
                                                     nullptr, nullptr, nullptr, 0, 0);
}

static void gemm_mix(hipStream_t s, const half_t* A, int lda, const half_t* Bt, int ldb,
                     const float* bias, int K, int M,
                     const float* wgp, const float* vpp, float* hvp, int e)
{
    dim3 g(M / 32, DMODEL / 64), b(256);
    gemm_kernel<ACT_NONE, EPI_MIX><<<g, b, 0, s>>>(A, lda, Bt, ldb, bias, K,
                                                   nullptr, 0, nullptr, nullptr, 1,
                                                   wgp, vpp, hvp, e, M);
}

// ============================================================================
extern "C" void kernel_launch(void* const* d_in, const int* in_sizes, int n_in,
                              void* d_out, int out_size, void* d_ws, size_t ws_size,
                              hipStream_t stream)
{
    (void)in_sizes; (void)n_in; (void)out_size; (void)ws_size;

    const float* coords = (const float*)d_in[0];
    const float* ctx    = (const float*)d_in[1];
    const float* eb0 = (const float*)d_in[27];
    const float* eb1 = (const float*)d_in[29];
    const float* eb2 = (const float*)d_in[31];
    const float* eb3 = (const float*)d_in[33];

    char* ws = (char*)d_ws;
    const size_t MB = 1024ull * 1024ull;
    // workspace plan (phase-overlapped; peak ~565 MB):
    const size_t WS_WT = 0;          // 16 MB: f16 transposed weights
    const size_t WS_WG = 16 * MB;    //  4 MB: top-2 gate weights f32 [V][N][8]
    const size_t WS_VP = 20 * MB;    //  1 MB: view_emb @ view_proj f32 [V][256]
    const size_t WS_HV = 21 * MB;    // 96 MB: hv f32 [V][N][256]
    const size_t WS_TOK = 117 * MB;  //128 MB: tokens f32 [N*4][256]
    const size_t WS_B1 = 245 * MB;   // 64 MB: gin / ynorm
    const size_t WS_B2 = 309 * MB;   // 64 MB: gh1 / eh1 / q / t1 / decin
    const size_t WS_B3 = 373 * MB;   // 64 MB: gh2 / eh2 / k / dec h1
    const size_t WS_B4 = 437 * MB;   // 64 MB: pe / v / dec h2
    const size_t WS_B5 = 501 * MB;   // 64 MB: attention out

    float* wg  = (float*)(ws + WS_WG);
    float* vp  = (float*)(ws + WS_VP);
    float* hv  = (float*)(ws + WS_HV);
    float* tok = (float*)(ws + WS_TOK);
    half_t* B1 = (half_t*)(ws + WS_B1);
    half_t* B2 = (half_t*)(ws + WS_B2);
    half_t* B3 = (half_t*)(ws + WS_B3);
    half_t* B4 = (half_t*)(ws + WS_B4);
    half_t* B5 = (half_t*)(ws + WS_B5);

    // ---- weight prep (f32 [K][N] -> f16 [N][Kp], zero-padded K) ----
    size_t wtoff = 0;
    auto walloc = [&](size_t elems) -> half_t* {
        half_t* p = (half_t*)(ws + WS_WT + wtoff);
        wtoff += ((elems * sizeof(half_t) + 255) & ~(size_t)255);
        return p;
    };
    auto prep = [&](const float* src, int K, int N, int Kp) -> half_t* {
        half_t* dst = walloc((size_t)N * Kp);
        int total = N * Kp;
        prep_kernel<<<(total + 255) / 256, 256, 0, stream>>>(src, dst, K, N, Kp);
        return dst;
    };

    half_t *eW0t[8], *eW1t[8], *eW2t[8], *eW3t[8];
    {
        const float* eW0 = (const float*)d_in[26];
        const float* eW1 = (const float*)d_in[28];
        const float* eW2 = (const float*)d_in[30];
        const float* eW3 = (const float*)d_in[32];
        for (int e = 0; e < NEXP; ++e) {
            eW0t[e] = prep(eW0 + (size_t)e * 39 * 256, 39, 256, 64);
            eW1t[e] = prep(eW1 + (size_t)e * 256 * 256, 256, 256, 256);
            eW2t[e] = prep(eW2 + (size_t)e * 256 * 256, 256, 256, 256);
            eW3t[e] = prep(eW3 + (size_t)e * 256 * 256, 256, 256, 256);
        }
    }
    half_t* gW0t = prep((const float*)d_in[66], 19, 128, 32);
    half_t* gW1t = prep((const float*)d_in[68], 128, 128, 128);
    half_t* gW2t = prep((const float*)d_in[70], 128, 128, 128);

    half_t *wqT[2], *wkT[2], *wvT[2], *woT[2], *w1T[2], *w2T[2];
    for (int L = 0; L < 2; ++L) {
        int base = 34 + L * 16; // ln1 ln2 w1 w2 wk wo wq wv
        w1T[L] = prep((const float*)d_in[base + 4], 256, 1024, 256);
        w2T[L] = prep((const float*)d_in[base + 6], 1024, 256, 1024);
        wkT[L] = prep((const float*)d_in[base + 8], 256, 256, 256);
        woT[L] = prep((const float*)d_in[base + 10], 256, 256, 256);
        wqT[L] = prep((const float*)d_in[base + 12], 256, 256, 256);
        wvT[L] = prep((const float*)d_in[base + 14], 256, 256, 256);
    }
    half_t *dW0t[3], *dW1t[3], *dW2t[3];
    for (int i = 0; i < 3; ++i) {
        int base = 2 + i * 8;
        dW0t[i] = prep((const float*)d_in[base + 0], 512, 256, 512);
        dW1t[i] = prep((const float*)d_in[base + 2], 256, 256, 256);
        dW2t[i] = prep((const float*)d_in[base + 4], 256, 256, 256);
    }

    // ---- gate: siren(19->128->128->128->8), softmax, top-2, renorm ----
    const int VN = NVIEW * NPTS;
    half_t* gin = B1; half_t* gh1 = B2; half_t* gh2 = B3;
    gin_kernel<<<(VN * 32) / 256, 256, 0, stream>>>(coords, (const float*)d_in[74], gin);
    gemm_f16<ACT_SIN>(stream, gin, 32, gW0t, 32, (const float*)d_in[67], 32, VN, 128, gh1, 128);
    gemm_f16<ACT_SIN>(stream, gh1, 128, gW1t, 128, (const float*)d_in[69], 128, VN, 128, gh2, 128);
    gemm_f16<ACT_SIN>(stream, gh2, 128, gW2t, 128, (const float*)d_in[71], 128, VN, 128, gh1, 128);
    gate_final_kernel<<<(VN + 127) / 128, 128, 0, stream>>>(
        gh1, (const float*)d_in[72], (const float*)d_in[73], wg, VN);

    // ---- experts + MoE mix into hv (with view_proj folded at e==0) ----
    half_t* pe = B4; half_t* eh1 = B2; half_t* eh2 = B3;
    pe_kernel<<<(NPTS * 64) / 256, 256, 0, stream>>>(coords, pe);
    vp_kernel<<<(NVIEW * DMODEL + 255) / 256, 256, 0, stream>>>(
        (const float*)d_in[74], (const float*)d_in[75], vp);
    for (int e = 0; e < NEXP; ++e) {
        gemm_f16<ACT_SIN>(stream, pe, 64, eW0t[e], 64, eb0 + e * 256, 64, NPTS, 256, eh1, 256);
        gemm_f16<ACT_SIN>(stream, eh1, 256, eW1t[e], 256, eb1 + e * 256, 256, NPTS, 256, eh2, 256);
        gemm_f16<ACT_SIN>(stream, eh2, 256, eW2t[e], 256, eb2 + e * 256, 256, NPTS, 256, eh1, 256);
        gemm_mix(stream, eh1, 256, eW3t[e], 256, eb3 + e * 256, 256, NPTS, wg, vp, hv, e);
    }

    // ---- assemble tokens [N][4][256]: ctx + 3 views ----
    tokens_kernel<<<NPTS * 4, 256, 0, stream>>>(hv, ctx, tok);

    // ---- fusion transformer x2 ----
    const int TR = NPTS * 4;
    half_t* ynorm = B1; half_t* qb = B2; half_t* kb = B3; half_t* vb = B4;
    half_t* ob = B5;    half_t* t1 = B2; // t1 reuses q after attention
    for (int L = 0; L < 2; ++L) {
        int base = 34 + L * 16;
        ln_kernel<<<TR / 8, 256, 0, stream>>>(tok, (const float*)d_in[base + 0],
                                              (const float*)d_in[base + 1], ynorm, TR);
        gemm_f16<ACT_NONE>(stream, ynorm, 256, wqT[L], 256, (const float*)d_in[base + 13],
                           256, TR, 256, qb, 256);
        gemm_f16<ACT_NONE>(stream, ynorm, 256, wkT[L], 256, (const float*)d_in[base + 9],
                           256, TR, 256, kb, 256);
        gemm_f16<ACT_NONE>(stream, ynorm, 256, wvT[L], 256, (const float*)d_in[base + 15],
                           256, TR, 256, vb, 256);
        attn_kernel<<<(NPTS * 32) / 256, 256, 0, stream>>>(qb, kb, vb, ob);
        gemm_resid(stream, ob, 256, woT[L], 256, (const float*)d_in[base + 11],
                   256, TR, tok, tok, 1);
        ln_kernel<<<TR / 8, 256, 0, stream>>>(tok, (const float*)d_in[base + 2],
                                              (const float*)d_in[base + 3], ynorm, TR);
        // FF: K-split into 4 chunks of 256 to keep t1 at 64 MB (L2-resident)
        for (int cch = 0; cch < 4; ++cch) {
            gemm_f16<ACT_GELU>(stream, ynorm, 256, w1T[L] + (size_t)cch * 256 * 256, 256,
                               (const float*)d_in[base + 5] + cch * 256, 256, TR, 256, t1, 256);
            gemm_resid(stream, t1, 256, w2T[L] + (size_t)cch * 256, 1024,
                       (const float*)d_in[base + 7], 256, TR, tok, tok, (cch == 3) ? 1 : 0);
        }
    }

    // ---- decoders ----
    half_t* din = B2; half_t* dh1 = B3; half_t* dh2 = B4;
    float* out = (float*)d_out;
    for (int i = 0; i < 3; ++i) {
        int base = 2 + i * 8;
        decin_kernel<<<(NPTS * 512) / 256, 256, 0, stream>>>(tok, din, i);
        gemm_f16<ACT_SIN>(stream, din, 512, dW0t[i], 512, (const float*)d_in[base + 1],
                          512, NPTS, 256, dh1, 256);
        gemm_f16<ACT_SIN>(stream, dh1, 256, dW1t[i], 256, (const float*)d_in[base + 3],
                          256, NPTS, 256, dh2, 256);
        gemm_f16<ACT_SIN>(stream, dh2, 256, dW2t[i], 256, (const float*)d_in[base + 5],
                          256, NPTS, 256, dh1, 256);
        if (i == 0)
            dec_final_kernel<3><<<NPTS / 256, 256, 0, stream>>>(
                dh1, (const float*)d_in[base + 6], (const float*)d_in[base + 7], out, 0, NPTS);
        else if (i == 1)
            dec_final_kernel<1><<<NPTS / 256, 256, 0, stream>>>(
                dh1, (const float*)d_in[base + 6], (const float*)d_in[base + 7], out, 3, NPTS);
        else
            dec_final_kernel<16><<<NPTS / 256, 256, 0, stream>>>(
                dh1, (const float*)d_in[base + 6], (const float*)d_in[base + 7], out, 4, NPTS);
    }
}
